// Vertical_attention_module_45183055954278
// MI455X (gfx1250) — compile-verified, matrix operand order checked by reading, not run on hardware
//
#include <hip/hip_runtime.h>
#include <hip/hip_bf16.h>

typedef __bf16 bf16;
typedef __attribute__((ext_vector_type(16))) __bf16 v16bf;
typedef __attribute__((ext_vector_type(8)))  float  v8f;

constexpr int BB  = 4;     // batch
constexpr int C   = 64;    // channels
constexpr int KC  = 32;    // qk channels (== WMMA K for bf16!)
constexpr int H   = 256;   // height (attention length)
constexpr int W   = 512;   // width (independent columns)
constexpr int HT  = 16;    // h-tile (WMMA M/N)
constexpr int NHT = H / HT;           // 16 tiles of 16
constexpr int WAVES = 8;
constexpr int THREADS = WAVES * 32;
constexpr float EPS = 1e-5f;

// ---- LDS layout (dynamic, 128 KB total; WGP has 320 KB) ----
constexpr int QS_OFF = 0;                    // qs[h][kc]  bf16  16 KB
constexpr int KS_OFF = QS_OFF + H * KC * 2;  // ks[g][kc]  bf16  16 KB
constexpr int VS_OFF = KS_OFF + H * KC * 2;  // vs[d][g]   bf16  32 KB
constexpr int UN_OFF = VS_OFF + C * H * 2;   // union: xcol f32 [c][h] (64 KB) | ps bf16 [wave][16][g] (64 KB)
constexpr int SMEM_BYTES = UN_OFF + C * H * 4;  // = 131072

// K index inside a 16-bit A/B fragment: VGPR j holds the packed pair (K, K+1)
__device__ __forceinline__ int frag_k(int j, int lane) {
  int kb = (lane >> 4) << 3;                       // lanes 16..31 handle K+8 / K+24 halves
  return ((j < 4) ? (2 * j) : (16 + 2 * (j - 4))) + kb;
}

__global__ void __launch_bounds__(THREADS)
vattn_kernel(const float* __restrict__ x,
             const float* __restrict__ Wq,
             const float* __restrict__ bnq_g, const float* __restrict__ bnq_b,
             const float* __restrict__ bnq_m, const float* __restrict__ bnq_v,
             const float* __restrict__ Wk,
             const float* __restrict__ bnk_g, const float* __restrict__ bnk_b,
             const float* __restrict__ bnk_m, const float* __restrict__ bnk_v,
             const float* __restrict__ Wv, const float* __restrict__ bv,
             const float* __restrict__ gamma,
             float* __restrict__ out)
{
  extern __shared__ char smem[];
  bf16*  qs   = (bf16*)(smem + QS_OFF);
  bf16*  ks   = (bf16*)(smem + KS_OFF);
  bf16*  vs   = (bf16*)(smem + VS_OFF);
  float* xcol = (float*)(smem + UN_OFF);
  bf16*  ps   = (bf16*)(smem + UN_OFF);   // reuses xcol space after phase 1

  const int w    = blockIdx.x;            // consecutive blocks -> consecutive w: L2 merges lines
  const int b    = blockIdx.y;
  const int tid  = threadIdx.x;
  const int lane = tid & 31;
  const int wave = tid >> 5;

  // ---------- Phase 0: stage x[:, :, w] column into LDS ----------
  {
    const float* xb = x + ((size_t)b * C) * H * W + w;
    for (int c = 0; c < C; ++c)
      xcol[c * H + tid] = xb[((size_t)c * H + tid) * W];
  }
  __syncthreads();

  // ---------- Phase 1: projections + BN/bias, f32 VALU, bf16 to LDS ----------
  {
    const int h = tid;                    // one height row per thread
    float xr[C];
    #pragma unroll
    for (int c = 0; c < C; ++c) xr[c] = xcol[c * H + h];

    for (int k = 0; k < KC; ++k) {        // W* indices uniform -> scalar loads
      float aq = 0.f, ak = 0.f;
      #pragma unroll
      for (int c = 0; c < C; ++c) {
        aq = fmaf(Wq[k * C + c], xr[c], aq);
        ak = fmaf(Wk[k * C + c], xr[c], ak);
      }
      float sq = bnq_g[k] * rsqrtf(bnq_v[k] + EPS);
      float sk = bnk_g[k] * rsqrtf(bnk_v[k] + EPS);
      qs[h * KC + k] = (bf16)(aq * sq + (bnq_b[k] - bnq_m[k] * sq));
      ks[h * KC + k] = (bf16)(ak * sk + (bnk_b[k] - bnk_m[k] * sk));
    }
    for (int d = 0; d < C; ++d) {
      float av = 0.f;
      #pragma unroll
      for (int c = 0; c < C; ++c) av = fmaf(Wv[d * C + c], xr[c], av);
      vs[d * H + h] = (bf16)(av + bv[d]);
    }
  }
  __syncthreads();                        // q/k/v ready; xcol dead -> ps may reuse it

  const float g0 = gamma[0];

  // ---------- Phase 2: per-wave attention over 16-row h-tiles ----------
  for (int ht = wave; ht < NHT; ht += WAVES) {
    const int h0 = ht * HT;
    const int m  = lane & 15;

    // Q A-fragment: A[M=h, K=kc], kc == 32 -> one fragment covers full contraction
    v16bf aq;
    {
      const bf16* qrow = qs + (h0 + m) * KC;
      #pragma unroll
      for (int j = 0; j < 8; ++j) {
        int kk = frag_k(j, lane);
        aq[2 * j]     = qrow[kk];
        aq[2 * j + 1] = qrow[kk + 1];
      }
    }

    // S[h-tile, 0..255] = Q^T K  (16 WMMAs, full row block held in 128 VGPRs/lane)
    v8f S[NHT];
    #pragma unroll
    for (int gt = 0; gt < NHT; ++gt) {
      v16bf bk;
      const bf16* krow = ks + (gt * 16 + m) * KC;   // B[K=kc, N=g]
      #pragma unroll
      for (int j = 0; j < 8; ++j) {
        int kk = frag_k(j, lane);
        bk[2 * j]     = krow[kk];
        bk[2 * j + 1] = krow[kk + 1];
      }
      v8f z = {0.f, 0.f, 0.f, 0.f, 0.f, 0.f, 0.f, 0.f};
      S[gt] = __builtin_amdgcn_wmma_f32_16x16x32_bf16(false, aq, false, bk,
                                                      (short)0, z, false, false);
    }

    // Softmax over g: row M = r + 8*(lane>=16) lives across a 16-lane group
    #pragma unroll
    for (int r = 0; r < 8; ++r) {
      float mx = -3.402823466e+38f;
      #pragma unroll
      for (int gt = 0; gt < NHT; ++gt) mx = fmaxf(mx, S[gt][r]);
      #pragma unroll
      for (int off = 1; off < 16; off <<= 1) mx = fmaxf(mx, __shfl_xor(mx, off, 32));
      float sum = 0.f;
      #pragma unroll
      for (int gt = 0; gt < NHT; ++gt) {
        float e = __expf(S[gt][r] - mx);
        S[gt][r] = e;
        sum += e;
      }
      #pragma unroll
      for (int off = 1; off < 16; off <<= 1) sum += __shfl_xor(sum, off, 32);
      float inv = 1.f / sum;
      #pragma unroll
      for (int gt = 0; gt < NHT; ++gt) S[gt][r] *= inv;
    }

    // Spill P (C-layout) to this wave's private LDS slab, re-read as A-fragments.
    bf16* pw = ps + wave * (HT * H);
    #pragma unroll
    for (int gt = 0; gt < NHT; ++gt) {
      int gg = gt * 16 + m;
      #pragma unroll
      for (int r = 0; r < 8; ++r) {
        int M = r + ((lane >> 4) << 3);
        pw[M * H + gg] = (bf16)S[gt][r];
      }
    }
    // same-wave LDS ops are in-order; drain DScnt before cross-lane re-read
    asm volatile("s_wait_dscnt 0" ::: "memory");

    // O[16h, 64d] = P (16x256) @ V^T (256x64): 8 K-steps x 4 d-tiles = 32 WMMAs.
    // Fully unrolled so the scheduler can batch ds_load_b128s across K-steps
    // instead of a per-iteration load->wait->wmma serialization. S is dead here,
    // so register pressure is fine.
    v8f O[4];
    #pragma unroll
    for (int dt = 0; dt < 4; ++dt)
      O[dt] = (v8f){0.f, 0.f, 0.f, 0.f, 0.f, 0.f, 0.f, 0.f};

    #pragma unroll
    for (int gs = 0; gs < 8; ++gs) {
      v16bf ap;
      const bf16* prow = pw + m * H + gs * 32;      // A[M=h, K=g]
      #pragma unroll
      for (int j = 0; j < 8; ++j) {
        int kk = frag_k(j, lane);
        ap[2 * j]     = prow[kk];
        ap[2 * j + 1] = prow[kk + 1];
      }
      #pragma unroll
      for (int dt = 0; dt < 4; ++dt) {
        v16bf bvf;
        const bf16* vrow = vs + (dt * 16 + m) * H + gs * 32;  // B[K=g, N=d] = v[d][g]
        #pragma unroll
        for (int j = 0; j < 8; ++j) {
          int kk = frag_k(j, lane);
          bvf[2 * j]     = vrow[kk];
          bvf[2 * j + 1] = vrow[kk + 1];
        }
        O[dt] = __builtin_amdgcn_wmma_f32_16x16x32_bf16(false, ap, false, bvf,
                                                        (short)0, O[dt], false, false);
      }
    }

    // Epilogue: out = gamma * O + x  (residual re-read hits L2)
    #pragma unroll
    for (int dt = 0; dt < 4; ++dt) {
      int d = dt * 16 + m;
      #pragma unroll
      for (int r = 0; r < 8; ++r) {
        int hh = h0 + r + ((lane >> 4) << 3);
        size_t idx = (((size_t)b * C + d) * H + hh) * W + w;
        out[idx] = g0 * O[dt][r] + x[idx];
      }
    }
  }
}

extern "C" void kernel_launch(void* const* d_in, const int* in_sizes, int n_in,
                              void* d_out, int out_size, void* d_ws, size_t ws_size,
                              hipStream_t stream) {
  (void)in_sizes; (void)n_in; (void)d_ws; (void)ws_size; (void)out_size;
  const float* x     = (const float*)d_in[0];
  const float* Wq    = (const float*)d_in[1];
  const float* bnq_g = (const float*)d_in[2];
  const float* bnq_b = (const float*)d_in[3];
  const float* bnq_m = (const float*)d_in[4];
  const float* bnq_v = (const float*)d_in[5];
  const float* Wk    = (const float*)d_in[6];
  const float* bnk_g = (const float*)d_in[7];
  const float* bnk_b = (const float*)d_in[8];
  const float* bnk_m = (const float*)d_in[9];
  const float* bnk_v = (const float*)d_in[10];
  const float* Wv    = (const float*)d_in[11];
  const float* bv    = (const float*)d_in[12];
  const float* gamma = (const float*)d_in[13];
  float* out = (float*)d_out;

  dim3 grid(W, BB);   // one workgroup per (w, b) column
  vattn_kernel<<<grid, dim3(THREADS), SMEM_BYTES, stream>>>(
      x, Wq, bnq_g, bnq_b, bnq_m, bnq_v,
      Wk, bnk_g, bnk_b, bnk_m, bnk_v,
      Wv, bv, gamma, out);
}
